// MultiHeadSelfAttention_43207370997830
// MI455X (gfx1250) — matrix operand order repaired and checked by reading, not compiled, not run on hardware
//
#include <hip/hip_runtime.h>

// ---------------------------------------------------------------------------
// MI455X (gfx1250) multi-head self-attention: bf16 WMMA + TDM-staged GEMMs +
// flash softmax.  B=4, N=2048, F=1024, H=16, DH=64.  SCALE = F/19.5.
// ---------------------------------------------------------------------------

#define B_DIM  4
#define N_DIM  2048
#define F_DIM  1024
#define H_DIM  16
#define DH_DIM 64

typedef __bf16 bf16_t;
typedef __attribute__((ext_vector_type(8)))  bf16_t       v8bf;
typedef __attribute__((ext_vector_type(16))) bf16_t       v16bf;
typedef __attribute__((ext_vector_type(8)))  float        v8f;
typedef __attribute__((ext_vector_type(4)))  unsigned int u32x4;
typedef __attribute__((ext_vector_type(4)))  int          i32x4;
typedef __attribute__((ext_vector_type(8)))  int          i32x8;

union BF16x16 { v16bf v; v8bf h[2]; bf16_t e[16]; };

#if defined(__has_builtin)
#if __has_builtin(__builtin_amdgcn_tensor_load_to_lds)
#define USE_TDM 1
#endif
#endif

// ---- WMMA fragment loaders (layouts per CDNA5 ISA 7.12.2, wave32) ----------
// A 16x32 bf16: lane L -> row m=L&15; j<8: k=8*half+j ; j>=8: k=16+8*half+(j-8)
__device__ __forceinline__ v16bf load_frag_a(const bf16_t* src, int ld) {
    const int lane = threadIdx.x & 31;
    const int m    = lane & 15;
    const int half = lane >> 4;
    const bf16_t* p = src + (size_t)m * ld + 8 * half;
    BF16x16 f;
    f.h[0] = *(const v8bf*)(p);        // k = 8*half .. +7
    f.h[1] = *(const v8bf*)(p + 16);   // k = 16+8*half .. +7
    return f.v;
}

// B 32x16 bf16 from Bt stored [n][k] row-major: lane -> col n=L&15,
// j<8: k=8*half+j ; j>=8: k=16+8*half+(j-8).
__device__ __forceinline__ v16bf load_frag_b(const bf16_t* srcT, int ld) {
    const int lane = threadIdx.x & 31;
    const int n    = lane & 15;
    const int half = lane >> 4;
    const bf16_t* p = srcT + (size_t)n * ld + 8 * half;
    BF16x16 f;
    f.h[0] = *(const v8bf*)(p);
    f.h[1] = *(const v8bf*)(p + 16);
    return f.v;
}

__device__ __forceinline__ v8f wmma_bf16(v16bf a, v16bf b, v8f c) {
    return __builtin_amdgcn_wmma_f32_16x16x32_bf16(false, a, false, b, (short)0, c,
                                                   false, false);
}

// ---- f32 -> bf16 cast ------------------------------------------------------
__global__ __launch_bounds__(256) void cvt_kernel(const float* __restrict__ src,
                                                  bf16_t* __restrict__ dst, int n) {
    int i = (blockIdx.x * blockDim.x + threadIdx.x) * 4;
    if (i < n) {
        float4 f = *(const float4*)(src + i);
        dst[i + 0] = (bf16_t)f.x;
        dst[i + 1] = (bf16_t)f.y;
        dst[i + 2] = (bf16_t)f.z;
        dst[i + 3] = (bf16_t)f.w;
    }
}

// ---- Staged GEMM: y = A @ W^T + b ------------------------------------------
// Block = 8 waves; all waves share one 32-row A tile staged to LDS by the TDM.
// Per wave: 32(M) x 64(N) tile, 8 wmma per K=32 step, register-double-buffered.
// mode 0/1: scatter bf16 to [b,h,n,dh] (Q,K); mode 2: [b,h,dh,n] (Vt);
// mode 3: f32 row-major output (final projection).
__global__ __launch_bounds__(256) void gemm_kernel(const bf16_t* __restrict__ abuf,
                                                   const bf16_t* __restrict__ W,
                                                   const float*  __restrict__ bias,
                                                   void* __restrict__ outp, int mode) {
    // +64 elements padding: the pipelined prefetch reads one step past k=F.
    __shared__ __align__(16) bf16_t atile[32 * F_DIM + 64];

    const int wave = threadIdx.x >> 5;
    const int lane = threadIdx.x & 31;
    const int row0 = (blockIdx.x >> 1) * 32;                    // 256 m-tiles
    const int col0 = ((blockIdx.x & 1) * 8 + wave) * 64;        // 16 n-tiles

    // ---- stage A tile (32 x 1024 bf16 = 64 KB) into LDS ----
#ifdef USE_TDM
    if (threadIdx.x < 32) {   // one wave issues the tensor DMA
        const unsigned long long ga =
            (unsigned long long)(uintptr_t)(abuf + (size_t)row0 * F_DIM);
        const unsigned la = (unsigned)(uintptr_t)(&atile[0]);
        u32x4 g0 = { 1u,                       // count=1, user descriptor
                     la,                       // lds_addr
                     (unsigned)ga,             // global_addr[31:0]
                     (unsigned)((ga >> 32) & 0x1FFFFFFu) | (2u << 30) }; // type=2
        i32x8 g1 = { (int)(1u << 16),          // data_size=1 (2B), no multicast
                     (int)(F_DIM << 16),       // tensor_dim0[15:0]
                     (int)(32u << 16),         // tensor_dim1 = 32 rows
                     (int)(F_DIM << 16),       // tile_dim0 = 1024
                     32,                       // tile_dim1 = 32
                     F_DIM,                    // tensor_dim0_stride
                     0, 0 };
        i32x4 gz = { 0, 0, 0, 0 };
#if __clang_major__ >= 23
        i32x8 gz8 = { 0, 0, 0, 0, 0, 0, 0, 0 };
        __builtin_amdgcn_tensor_load_to_lds(g0, g1, gz, gz, gz8, 0);
#else
        __builtin_amdgcn_tensor_load_to_lds(g0, g1, gz, gz, 0);
#endif
#if __has_builtin(__builtin_amdgcn_s_wait_tensorcnt)
        __builtin_amdgcn_s_wait_tensorcnt(0);
#else
        asm volatile("s_wait_tensorcnt 0x0" ::: "memory");
#endif
    }
#else
    for (int it = threadIdx.x; it < 32 * (F_DIM / 8); it += 256) {
        const int row = it >> 7;            // 128 8-elem chunks per row
        const int c8  = it & 127;
        *(v8bf*)(atile + row * F_DIM + c8 * 8) =
            *(const v8bf*)(abuf + (size_t)(row0 + row) * F_DIM + c8 * 8);
    }
#endif
    __syncthreads();

    // ---- K loop: A frags from LDS, B frags from global (L2-resident W) ----
    v8f acc[2][4] = {};
    v16bf a0 = load_frag_a(atile + 0 * F_DIM, F_DIM);
    v16bf a1 = load_frag_a(atile + 16 * F_DIM, F_DIM);
    v16bf b[4];
#pragma unroll
    for (int t = 0; t < 4; ++t)
        b[t] = load_frag_b(W + (size_t)(col0 + 16 * t) * F_DIM, F_DIM);

#pragma unroll 2
    for (int k = 0; k < F_DIM; k += 32) {
        // prefetch next K step (last one reads padded/adjacent junk, unused)
        v16bf na0 = load_frag_a(atile + 0 * F_DIM + k + 32, F_DIM);
        v16bf na1 = load_frag_a(atile + 16 * F_DIM + k + 32, F_DIM);
        v16bf nb[4];
#pragma unroll
        for (int t = 0; t < 4; ++t)
            nb[t] = load_frag_b(W + (size_t)(col0 + 16 * t) * F_DIM + k + 32, F_DIM);
#pragma unroll
        for (int t = 0; t < 4; ++t) {
            acc[0][t] = wmma_bf16(a0, b[t], acc[0][t]);
            acc[1][t] = wmma_bf16(a1, b[t], acc[1][t]);
        }
        a0 = na0; a1 = na1;
#pragma unroll
        for (int t = 0; t < 4; ++t) b[t] = nb[t];
    }

    // ---- epilogue: bias + scatter ----
    const int n    = lane & 15;
    const int half = lane >> 4;
#pragma unroll
    for (int t = 0; t < 4; ++t) {
        const int col = col0 + 16 * t + n;
        const float bv = bias[col];
        const int h = col >> 6;
        const int d = col & 63;
#pragma unroll
        for (int ms = 0; ms < 2; ++ms) {
#pragma unroll
            for (int r = 0; r < 8; ++r) {
                const int row  = row0 + 16 * ms + r + 8 * half;  // global M
                const float val = acc[ms][t][r] + bv;
                if (mode == 3) {
                    ((float*)outp)[(size_t)row * F_DIM + col] = val;
                } else {
                    const int bidx = row >> 11;
                    const int nseq = row & (N_DIM - 1);
                    size_t o;
                    if (mode == 2)
                        o = ((size_t)(bidx * H_DIM + h) * DH_DIM + d) * N_DIM + nseq;
                    else
                        o = ((size_t)(bidx * H_DIM + h) * N_DIM + nseq) * DH_DIM + d;
                    ((bf16_t*)outp)[o] = (bf16_t)val;
                }
            }
        }
    }
}

// ---- Flash attention: one wave per (b,h, 16 query rows) --------------------
__global__ __launch_bounds__(32) void attn_kernel(const bf16_t* __restrict__ Qb,
                                                  const bf16_t* __restrict__ Kb,
                                                  const bf16_t* __restrict__ Vtb,
                                                  bf16_t* __restrict__ attnb) {
    __shared__ __align__(16) bf16_t pbuf[16 * 32];   // P transpose bounce

    const int lane = threadIdx.x & 31;
    const int qt   = blockIdx.x & (N_DIM / 16 - 1);  // 0..127
    const int bh   = blockIdx.x >> 7;                // 0..63

    const bf16_t* Q  = Qb  + (size_t)bh * N_DIM * DH_DIM + (size_t)qt * 16 * DH_DIM;
    const bf16_t* K  = Kb  + (size_t)bh * N_DIM * DH_DIM;
    const bf16_t* Vt = Vtb + (size_t)bh * DH_DIM * N_DIM;

    const v16bf qa0 = load_frag_a(Q + 0,  DH_DIM);
    const v16bf qa1 = load_frag_a(Q + 32, DH_DIM);

    const float inv_scale = 19.5f / (float)F_DIM;
    const int   half = lane >> 4;
    const int   n    = lane & 15;

    float mrow[8], lrow[8];
#pragma unroll
    for (int r = 0; r < 8; ++r) { mrow[r] = -1e30f; lrow[r] = 0.0f; }
    v8f oacc[4] = {};

    for (int j0 = 0; j0 < N_DIM; j0 += 32) {
        // prefetch next key/value chunk into caches (global_prefetch_b8)
        __builtin_prefetch(K  + (size_t)(j0 + 32) * DH_DIM + lane * 64, 0, 0);
        __builtin_prefetch(Vt + (size_t)lane        * N_DIM + j0 + 32, 0, 0);
        __builtin_prefetch(Vt + (size_t)(lane + 32) * N_DIM + j0 + 32, 0, 0);

        // ---- S = Q K^T for 32 keys ----
        v8f s[2] = {};
#pragma unroll
        for (int c = 0; c < 2; ++c) {
            v16bf b0 = load_frag_b(K + (size_t)(j0 + 16 * c) * DH_DIM + 0,  DH_DIM);
            s[c] = wmma_bf16(qa0, b0, s[c]);
            v16bf b1 = load_frag_b(K + (size_t)(j0 + 16 * c) * DH_DIM + 32, DH_DIM);
            s[c] = wmma_bf16(qa1, b1, s[c]);
        }
#pragma unroll
        for (int c = 0; c < 2; ++c)
#pragma unroll
            for (int r = 0; r < 8; ++r) s[c][r] *= inv_scale;

        // ---- online softmax: row stats across 16-lane half-groups ----
        float mnew[8], fac[8];
#pragma unroll
        for (int r = 0; r < 8; ++r) {
            float v = fmaxf(s[0][r], s[1][r]);
            v = fmaxf(v, __shfl_xor(v, 1, 32));
            v = fmaxf(v, __shfl_xor(v, 2, 32));
            v = fmaxf(v, __shfl_xor(v, 4, 32));
            v = fmaxf(v, __shfl_xor(v, 8, 32));
            mnew[r] = fmaxf(mrow[r], v);
            fac[r]  = __expf(mrow[r] - mnew[r]);
            mrow[r] = mnew[r];
        }
#pragma unroll
        for (int c = 0; c < 2; ++c)
#pragma unroll
            for (int r = 0; r < 8; ++r) s[c][r] = __expf(s[c][r] - mnew[r]);
#pragma unroll
        for (int r = 0; r < 8; ++r) {
            float v = s[0][r] + s[1][r];
            v += __shfl_xor(v, 1, 32);
            v += __shfl_xor(v, 2, 32);
            v += __shfl_xor(v, 4, 32);
            v += __shfl_xor(v, 8, 32);
            lrow[r] = lrow[r] * fac[r] + v;
        }
#pragma unroll
        for (int t = 0; t < 4; ++t)
#pragma unroll
            for (int r = 0; r < 8; ++r) oacc[t][r] *= fac[r];

        // ---- P: C-layout -> A-layout via LDS transpose ----
        __syncthreads();
#pragma unroll
        for (int c = 0; c < 2; ++c)
#pragma unroll
            for (int r = 0; r < 8; ++r)
                pbuf[(r + 8 * half) * 32 + c * 16 + n] = (bf16_t)s[c][r];
        __syncthreads();
        v16bf pa = load_frag_a(pbuf, 32);

        // ---- O += P V ----
#pragma unroll
        for (int t = 0; t < 4; ++t) {
            v16bf vb = load_frag_b(Vt + (size_t)(16 * t) * N_DIM + j0, N_DIM);
            oacc[t] = wmma_bf16(pa, vb, oacc[t]);
        }
    }

    // ---- finalize: O /= l, store as [b, n, h*64+d] bf16 ----
    const int h  = bh & (H_DIM - 1);
    const int bb = bh >> 4;
#pragma unroll
    for (int t = 0; t < 4; ++t) {
#pragma unroll
        for (int r = 0; r < 8; ++r) {
            const int seq = qt * 16 + r + 8 * half;
            float val = oacc[t][r] / lrow[r];
            size_t o = ((size_t)(bb * N_DIM + seq)) * F_DIM + h * DH_DIM + 16 * t + n;
            attnb[o] = (bf16_t)val;
        }
    }
}

// ---------------------------------------------------------------------------
extern "C" void kernel_launch(void* const* d_in, const int* in_sizes, int n_in,
                              void* d_out, int out_size, void* d_ws, size_t ws_size,
                              hipStream_t stream) {
    const float* x  = (const float*)d_in[0];
    const float* Wq = (const float*)d_in[1];
    const float* bq = (const float*)d_in[2];
    const float* Wk = (const float*)d_in[3];
    const float* bk = (const float*)d_in[4];
    const float* Wv = (const float*)d_in[5];
    const float* bv = (const float*)d_in[6];
    const float* Wo = (const float*)d_in[7];
    const float* bo = (const float*)d_in[8];
    float* out = (float*)d_out;

    const size_t MN = (size_t)B_DIM * N_DIM;  // 8192
    char* ws = (char*)d_ws;
    size_t off = 0;
    bf16_t* xb    = (bf16_t*)(ws + off); off += MN * F_DIM * 2;            // 16 MB
    bf16_t* wqb   = (bf16_t*)(ws + off); off += (size_t)F_DIM * F_DIM * 2; //  2 MB
    bf16_t* wkb   = (bf16_t*)(ws + off); off += (size_t)F_DIM * F_DIM * 2;
    bf16_t* wvb   = (bf16_t*)(ws + off); off += (size_t)F_DIM * F_DIM * 2;
    bf16_t* wob   = (bf16_t*)(ws + off); off += (size_t)F_DIM * F_DIM * 2;
    bf16_t* Qb    = (bf16_t*)(ws + off); off += MN * F_DIM * 2;            // 16 MB
    bf16_t* Kb    = (bf16_t*)(ws + off); off += MN * F_DIM * 2;
    bf16_t* Vtb   = (bf16_t*)(ws + off); off += MN * F_DIM * 2;
    bf16_t* attnb = (bf16_t*)(ws + off); off += MN * F_DIM * 2;            // 88 MB

    // 1) casts to bf16
    const int nx = (int)(MN * F_DIM);
    const int nw = F_DIM * F_DIM;
    cvt_kernel<<<nx / 4 / 256, 256, 0, stream>>>(x,  xb,  nx);
    cvt_kernel<<<nw / 4 / 256, 256, 0, stream>>>(Wq, wqb, nw);
    cvt_kernel<<<nw / 4 / 256, 256, 0, stream>>>(Wk, wkb, nw);
    cvt_kernel<<<nw / 4 / 256, 256, 0, stream>>>(Wv, wvb, nw);
    cvt_kernel<<<nw / 4 / 256, 256, 0, stream>>>(Wo, wob, nw);

    // 2) Q/K/V projections: 512 blocks x 8 waves, TDM-staged A tiles
    gemm_kernel<<<512, 256, 0, stream>>>(xb, wqb, bq, (void*)Qb,  0);
    gemm_kernel<<<512, 256, 0, stream>>>(xb, wkb, bk, (void*)Kb,  1);
    gemm_kernel<<<512, 256, 0, stream>>>(xb, wvb, bv, (void*)Vtb, 2);

    // 3) flash attention: B*H*(N/16) = 8192 single-wave blocks
    attn_kernel<<<B_DIM * H_DIM * (N_DIM / 16), 32, 0, stream>>>(Qb, Kb, Vtb, attnb);

    // 4) output projection (f32)
    gemm_kernel<<<512, 256, 0, stream>>>(attnb, wob, bo, (void*)out, 3);
}
